// TransformerBlock_78348793414197
// MI455X (gfx1250) — compile-verified, matrix operand order checked by reading, not run on hardware
//
#include <hip/hip_runtime.h>

typedef __bf16 bf16;
typedef __attribute__((ext_vector_type(16))) __bf16 v16bf;
typedef __attribute__((ext_vector_type(8)))  __bf16 v8bf;
typedef __attribute__((ext_vector_type(8)))  float  v8f;

__device__ __forceinline__ bf16 to_bf(float f) { return (bf16)f; }

__device__ __forceinline__ v8f wmma_bf16(v16bf a, v16bf b, v8f c) {
  return __builtin_amdgcn_wmma_f32_16x16x32_bf16(false, a, false, b, (short)0, c, false, false);
}

// ---- DPP16 butterfly reductions over 16-lane halves (no LDS traffic) ----
template <int CTRL>
__device__ __forceinline__ float dpp_f32(float x) {
  return __builtin_bit_cast(float,
      __builtin_amdgcn_mov_dpp(__builtin_bit_cast(int, x), CTRL, 0xF, 0xF, true));
}
// quad_perm(1,0,3,2)=0xB1 (xor1), quad_perm(2,3,0,1)=0x4E (xor2),
// row_half_mirror=0x141 (i<->7-i merges quads), row_mirror=0x140 (i<->15-i merges octets)
__device__ __forceinline__ float red16_max(float x) {
  x = fmaxf(x, dpp_f32<0xB1>(x));
  x = fmaxf(x, dpp_f32<0x4E>(x));
  x = fmaxf(x, dpp_f32<0x141>(x));
  x = fmaxf(x, dpp_f32<0x140>(x));
  return x;
}
__device__ __forceinline__ float red16_sum(float x) {
  x += dpp_f32<0xB1>(x);
  x += dpp_f32<0x4E>(x);
  x += dpp_f32<0x141>(x);
  x += dpp_f32<0x140>(x);
  return x;
}

// Load a 16x32 bf16 fragment in CDNA5 A/B per-lane layout from a row-major
// array with leading dimension ld (elements). Lanes 0-15: row r0+lane,
// K {c0+0..7, c0+16..23}; lanes 16-31: row r0+lane-16, K {c0+8..15, c0+24..31}.
__device__ __forceinline__ v16bf load_frag(const bf16* __restrict__ p, int ld, int r0, int c0) {
  const int lane = threadIdx.x & 31;
  const int half = lane >> 4;
  const bf16* q = p + (size_t)(r0 + (lane & 15)) * ld + c0 + half * 8;
  v8bf lo = *(const v8bf*)(q);
  v8bf hi = *(const v8bf*)(q + 16);
  return __builtin_shufflevector(lo, hi, 0,1,2,3,4,5,6,7,8,9,10,11,12,13,14,15);
}

// ---------------------------------------------------------------- layernorm
__global__ __launch_bounds__(256)
void ln_kernel(const float* __restrict__ x, const float* __restrict__ g,
               const float* __restrict__ be, bf16* __restrict__ out) {
  const int lane = threadIdx.x & 31;
  const int wid  = threadIdx.x >> 5;
  const size_t row = (size_t)blockIdx.x * 8 + wid;
  const float* xr = x + row * 384;
  float v[12];
  float s = 0.f;
#pragma unroll
  for (int i = 0; i < 12; ++i) { v[i] = xr[lane + i * 32]; s += v[i]; }
#pragma unroll
  for (int off = 16; off >= 1; off >>= 1) s += __shfl_xor(s, off, 32);
  const float mu = s * (1.0f / 384.0f);
  float q = 0.f;
#pragma unroll
  for (int i = 0; i < 12; ++i) { float d = v[i] - mu; q += d * d; }
#pragma unroll
  for (int off = 16; off >= 1; off >>= 1) q += __shfl_xor(q, off, 32);
  const float rstd = rsqrtf(q * (1.0f / 384.0f) + 1e-5f);
  bf16* orow = out + row * 384;
#pragma unroll
  for (int i = 0; i < 12; ++i) {
    const int c = lane + i * 32;
    orow[c] = to_bf((v[i] - mu) * rstd * g[c] + be[c]);
  }
}

// ------------------------------------------------- weight repack (fp32->bf16, transposed [N,K])
__global__ __launch_bounds__(256)
void repack_kernel(const float* __restrict__ Wq, const float* __restrict__ Wk,
                   const float* __restrict__ Wv, const float* __restrict__ Wp,
                   const float* __restrict__ W1, const float* __restrict__ W2,
                   bf16* __restrict__ wqkv, bf16* __restrict__ wp,
                   bf16* __restrict__ w1,   bf16* __restrict__ w2) {
  const int idx = blockIdx.x * 256 + threadIdx.x;
  const int n_qkv = 1152 * 384, n_p = 384 * 384, n_1 = 1536 * 384, n_2 = 384 * 1536;
  if (idx < n_qkv) {
    const int j = idx / 384, c = idx % 384;   // output col j of [C,1152]
    float v;
    if (j < 384)      { const int h = j >> 6,         d = j & 63;         v = Wq[((size_t)h * 384 + c) * 64 + d] * 0.125f; }
    else if (j < 768) { const int h = (j - 384) >> 6, d = (j - 384) & 63; v = Wk[((size_t)h * 384 + c) * 64 + d]; }
    else              { const int h = (j - 768) >> 6, d = (j - 768) & 63; v = Wv[((size_t)h * 384 + c) * 64 + d]; }
    wqkv[idx] = to_bf(v);
  } else if (idx < n_qkv + n_p) {
    const int k = idx - n_qkv, j = k / 384, c = k % 384;
    wp[k] = to_bf(Wp[(size_t)c * 384 + j]);
  } else if (idx < n_qkv + n_p + n_1) {
    const int k = idx - n_qkv - n_p, j = k / 384, c = k % 384;   // j: 0..1535
    w1[k] = to_bf(W1[(size_t)c * 1536 + j]);
  } else if (idx < n_qkv + n_p + n_1 + n_2) {
    const int k = idx - n_qkv - n_p - n_1, j = k / 1536, kk = k % 1536; // j: 0..383
    w2[k] = to_bf(W2[(size_t)kk * 384 + j]);
  }
}

// ---------------------------------------------------------------- tiled GEMM
// C[M,N] = A[M,K](bf16,row) * Bt[N,K](bf16,row = B col-major), f32 accumulate.
// 128x128 tile, K-tile 64, 8 waves of 32x64, register-double-buffered staging.
template <bool BIAS, bool RELU, bool RES, bool QKV>
__global__ __launch_bounds__(256)
void gemm_bf16(const bf16* __restrict__ A, const bf16* __restrict__ Bt,
               int M, int N, int K,
               const float* __restrict__ bias, const float* __restrict__ res,
               float* __restrict__ outf, bf16* __restrict__ outb,
               bf16* __restrict__ vt) {
  __shared__ bf16 sA[128 * 64];
  __shared__ bf16 sB[128 * 64];
  const int tid = threadIdx.x;
  const int lane = tid & 31;
  const int wid = tid >> 5;
  const int wm = wid & 3;   // 4 waves along M
  const int wn = wid >> 2;  // 2 waves along N
  const int bm = blockIdx.y * 128;
  const int bn = blockIdx.x * 128;

  v8f acc[2][4];
#pragma unroll
  for (int i = 0; i < 2; ++i)
#pragma unroll
    for (int j = 0; j < 4; ++j) acc[i][j] = (v8f)(0.0f);

  const int lr = tid >> 1;          // 0..127 staging row
  const int lc = (tid & 1) * 32;    // staging col base (32 bf16 per thread)
  const bf16* ga = A  + (size_t)(bm + lr) * K + lc;
  const bf16* gb = Bt + (size_t)(bn + lr) * K + lc;
  bf16* da = &sA[lr * 64 + lc];
  bf16* db = &sB[lr * 64 + lc];

  // prologue: fetch k-tile 0 into registers
  v8bf ar[4], br[4];
#pragma unroll
  for (int u = 0; u < 4; ++u) {
    ar[u] = *(const v8bf*)(ga + u * 8);
    br[u] = *(const v8bf*)(gb + u * 8);
  }

  for (int kt = 0; kt < K; kt += 64) {
    __syncthreads();
#pragma unroll
    for (int u = 0; u < 4; ++u) {
      *(v8bf*)(da + u * 8) = ar[u];
      *(v8bf*)(db + u * 8) = br[u];
    }
    __syncthreads();
    if (kt + 64 < K) {  // prefetch next k-tile while computing this one
#pragma unroll
      for (int u = 0; u < 4; ++u) {
        ar[u] = *(const v8bf*)(ga + kt + 64 + u * 8);
        br[u] = *(const v8bf*)(gb + kt + 64 + u * 8);
      }
    }
#pragma unroll
    for (int ks = 0; ks < 64; ks += 32) {
      v16bf af[2], bfr[4];
      af[0] = load_frag(sA, 64, wm * 32 + 0, ks);
      af[1] = load_frag(sA, 64, wm * 32 + 16, ks);
#pragma unroll
      for (int j = 0; j < 4; ++j) bfr[j] = load_frag(sB, 64, wn * 64 + j * 16, ks);
#pragma unroll
      for (int i = 0; i < 2; ++i)
#pragma unroll
        for (int j = 0; j < 4; ++j) acc[i][j] = wmma_bf16(af[i], bfr[j], acc[i][j]);
    }
  }

  const int half = lane >> 4;
  const int nl = lane & 15;
#pragma unroll
  for (int i = 0; i < 2; ++i) {
#pragma unroll
    for (int j = 0; j < 4; ++j) {
      const int m0 = bm + wm * 32 + i * 16;
      const int n0 = bn + wn * 64 + j * 16;
#pragma unroll
      for (int r = 0; r < 8; ++r) {
        const int gm = m0 + r + half * 8;
        const int gn = n0 + nl;
        float v = acc[i][j][r];
        if (BIAS) v += bias[gn];
        if (RELU) v = fmaxf(v, 0.0f);
        if (RES)  v += res[(size_t)gm * N + gn];
        if (outf) outf[(size_t)gm * N + gn] = v;
        if (outb) outb[(size_t)gm * N + gn] = to_bf(v);
        if (QKV) {
          if (gn >= 768) {  // V columns: also write V^T per (b,h): vt[(bh*64+d)*256 + t]
            const int b = gm >> 8, t = gm & 255;
            const int h = (gn - 768) >> 6, d = (gn - 768) & 63;
            vt[(((size_t)b * 6 + h) * 64 + d) * 256 + t] = to_bf(v);
          }
        }
      }
    }
  }
}

// ------------------------------------------------------------ flash attention
// One wave handles one (b,h) and one 16-row Q tile. T=256, d=64.
__global__ __launch_bounds__(256)
void attn_kernel(const bf16* __restrict__ qkv, const bf16* __restrict__ vt,
                 bf16* __restrict__ out) {
  __shared__ bf16 sP[8][16 * 32];
  const int lane = threadIdx.x & 31;
  const int wid = threadIdx.x >> 5;
  const int job = blockIdx.x * 8 + wid;  // 24576 jobs
  const int bh = job >> 4;
  const int qt = job & 15;
  const int b = bh / 6, h = bh % 6;
  const bf16* Qp = qkv + (size_t)(b * 256) * 1152 + h * 64;        // [256,1152] view
  const bf16* Kp = Qp + 384;
  const bf16* Vp = vt + (size_t)bh * 64 * 256;                     // [64,256] V^T
  bf16* sp = &sP[wid][0];

  const v16bf qa0 = load_frag(Qp, 1152, qt * 16, 0);
  const v16bf qa1 = load_frag(Qp, 1152, qt * 16, 32);

  const int half = lane >> 4;
  const int nl = lane & 15;
  float mrow[8], lrow[8];
  v8f O[4];
#pragma unroll
  for (int r = 0; r < 8; ++r) { mrow[r] = -__builtin_inff(); lrow[r] = 0.f; }
#pragma unroll
  for (int t = 0; t < 4; ++t) O[t] = (v8f)(0.0f);

  // software-pipelined K-fragment fetch
  v16bf kb0 = load_frag(Kp, 1152, 0, 0);
  v16bf kb1 = load_frag(Kp, 1152, 0, 32);

  for (int s = 0; s <= qt; ++s) {
    const v16bf ck0 = kb0, ck1 = kb1;
    if (s < qt) {
      kb0 = load_frag(Kp, 1152, (s + 1) * 16, 0);
      kb1 = load_frag(Kp, 1152, (s + 1) * 16, 32);
    }
    v8f sc = (v8f)(0.0f);
    sc = wmma_bf16(qa0, ck0, sc);
    sc = wmma_bf16(qa1, ck1, sc);
    if (s == qt) {  // causal mask on diagonal tile: valid iff m >= n
#pragma unroll
      for (int r = 0; r < 8; ++r) {
        const int m = r + half * 8;
        if (m < nl) sc[r] = -__builtin_inff();
      }
    }
    float p[8], alpha[8];
#pragma unroll
    for (int r = 0; r < 8; ++r) {
      const float mx = red16_max(sc[r]);
      const float mn = fmaxf(mrow[r], mx);
      alpha[r] = __expf(mrow[r] - mn);
      mrow[r] = mn;
      const float pe = __expf(sc[r] - mn);
      lrow[r] = lrow[r] * alpha[r] + red16_sum(pe);
      p[r] = pe;
    }
#pragma unroll
    for (int t = 0; t < 4; ++t)
#pragma unroll
      for (int r = 0; r < 8; ++r) O[t][r] *= alpha[r];

    const int cb = (s & 1) * 16;
#pragma unroll
    for (int r = 0; r < 8; ++r) sp[(r + half * 8) * 32 + cb + nl] = to_bf(p[r]);
    const bool lastodd = (s == qt) && ((s & 1) == 0);
    if (lastodd) {
#pragma unroll
      for (int r = 0; r < 8; ++r) sp[(r + half * 8) * 32 + 16 + nl] = to_bf(0.0f);
    }
    if ((s & 1) || lastodd) {
      asm volatile("s_wait_dscnt 0x0" ::: "memory");  // DS in-order; ensure staged P visible
      const v16bf pa = load_frag(sp, 32, 0, 0);
      const int c0 = (s & ~1) * 16;  // token base of this 32-wide chunk
#pragma unroll
      for (int t = 0; t < 4; ++t) {
        const v16bf vb = load_frag(Vp, 256, t * 16, c0);
        O[t] = wmma_bf16(pa, vb, O[t]);
      }
    }
  }

  const size_t obase = ((size_t)(b * 256 + qt * 16)) * 384 + h * 64;
#pragma unroll
  for (int r = 0; r < 8; ++r) {
    const float inv = 1.0f / lrow[r];
#pragma unroll
    for (int t = 0; t < 4; ++t)
      out[obase + (size_t)(r + half * 8) * 384 + t * 16 + nl] = to_bf(O[t][r] * inv);
  }
}

// ------------------------------------------------------------------ launcher
extern "C" void kernel_launch(void* const* d_in, const int* in_sizes, int n_in,
                              void* d_out, int out_size, void* d_ws, size_t ws_size,
                              hipStream_t stream) {
  (void)in_sizes; (void)n_in; (void)out_size; (void)ws_size;
  const float* x  = (const float*)d_in[0];
  const float* Wq = (const float*)d_in[1];
  const float* Wk = (const float*)d_in[2];
  const float* Wv = (const float*)d_in[3];
  const float* Wp = (const float*)d_in[4];
  const float* bp = (const float*)d_in[5];
  const float* W1 = (const float*)d_in[6];
  const float* b1 = (const float*)d_in[7];
  const float* W2 = (const float*)d_in[8];
  const float* b2 = (const float*)d_in[9];
  const float* g1 = (const float*)d_in[10];
  const float* be1 = (const float*)d_in[11];
  const float* g2 = (const float*)d_in[12];
  const float* be2 = (const float*)d_in[13];
  float* out = (float*)d_out;

  char* ws = (char*)d_ws;
  size_t o = 0;
  auto alloc = [&](size_t bytes) { size_t r = o; o = (o + bytes + 255) & ~(size_t)255; return r; };
  const size_t o_wqkv = alloc((size_t)1152 * 384 * 2);
  const size_t o_wp   = alloc((size_t)384 * 384 * 2);
  const size_t o_w1   = alloc((size_t)1536 * 384 * 2);
  const size_t o_w2   = alloc((size_t)384 * 1536 * 2);
  const size_t o_h    = alloc((size_t)65536 * 384 * 2);
  const size_t o_qkv  = alloc((size_t)65536 * 1152 * 2);  // 256B aligned already
  const size_t o_vt   = alloc((size_t)1536 * 64 * 256 * 2);
  const size_t o_attn = alloc((size_t)65536 * 384 * 2);
  const size_t o_ff   = o_qkv;  // ff [65536,1536] overlays qkv+vt exactly (attention done first)

  bf16* wqkvT = (bf16*)(ws + o_wqkv);
  bf16* wpT   = (bf16*)(ws + o_wp);
  bf16* w1T   = (bf16*)(ws + o_w1);
  bf16* w2T   = (bf16*)(ws + o_w2);
  bf16* hbuf  = (bf16*)(ws + o_h);
  bf16* qkvb  = (bf16*)(ws + o_qkv);
  bf16* vtb   = (bf16*)(ws + o_vt);
  bf16* attnb = (bf16*)(ws + o_attn);
  bf16* ffb   = (bf16*)(ws + o_ff);

  // 1) repack weights to bf16 [N,K]
  repack_kernel<<<6912, 256, 0, stream>>>(Wq, Wk, Wv, Wp, W1, W2, wqkvT, wpT, w1T, w2T);
  // 2) h = LN1(x)
  ln_kernel<<<8192, 256, 0, stream>>>(x, g1, be1, hbuf);
  // 3) QKV = h @ Wqkv (Q pre-scaled by 1/sqrt(64)); also emits V^T
  gemm_bf16<false, false, false, true><<<dim3(9, 512), 256, 0, stream>>>(
      hbuf, wqkvT, 65536, 1152, 384, nullptr, nullptr, nullptr, qkvb, vtb);
  // 4) flash attention -> attn_out (bf16)
  attn_kernel<<<3072, 256, 0, stream>>>(qkvb, vtb, attnb);
  // 5) x1 = x + attn_out @ Wp + bp  (f32, into d_out)
  gemm_bf16<true, false, true, false><<<dim3(3, 512), 256, 0, stream>>>(
      attnb, wpT, 65536, 384, 384, bp, x, out, nullptr, nullptr);
  // 6) h2 = LN2(x1)
  ln_kernel<<<8192, 256, 0, stream>>>(out, g2, be2, hbuf);
  // 7) ff = relu(h2 @ W1 + b1)  (bf16, overlays qkv/vt)
  gemm_bf16<true, true, false, false><<<dim3(12, 512), 256, 0, stream>>>(
      hbuf, w1T, 65536, 1536, 384, b1, nullptr, nullptr, ffb, nullptr);
  // 8) out = x1 + ff @ W2 + b2
  gemm_bf16<true, false, true, false><<<dim3(3, 512), 256, 0, stream>>>(
      ffb, w2T, 65536, 384, 1536, b2, out, out, nullptr, nullptr);
}
